// DCRNNEncoder_6640019440005
// MI455X (gfx1250) — compile-verified, matrix operand order checked by reading, not run on hardware
//
#include <hip/hip_runtime.h>
#include <hip/hip_bf16.h>

typedef __attribute__((ext_vector_type(16))) __bf16 v16bf;
typedef __attribute__((ext_vector_type(8)))  float  v8f;
typedef __attribute__((ext_vector_type(8)))  unsigned short usv8;

union FragBF { v16bf v; uint4 q[2]; };

__device__ __forceinline__ unsigned short f2bf(float f) {
    unsigned int u = __float_as_uint(f);
    u += 0x7FFFu + ((u >> 16) & 1u);          // round-to-nearest-even
    return (unsigned short)(u >> 16);
}
__device__ __forceinline__ float bf2f(unsigned short b) {
    return __uint_as_float(((unsigned int)b) << 16);
}

enum { MODE_DIFF1 = 0, MODE_DIFF2 = 1, MODE_GATE = 2, MODE_CAND = 3 };

#define LDS_STRIDE 40   // 32 K-elems + 8 pad (keeps 16B alignment, dodges bank conflicts)

// C[128x128] tile = A[128xK] * B[Kx128], bf16 in, f32 accum via v_wmma_f32_16x16x32_bf16.
// All dims are exact multiples of the tile (buffers are padded), so no guards.
// Global->LDS staging is register double-buffered: tile kt+1 is loaded into VGPRs
// while the WMMAs for tile kt run, so the loadcnt wait is hidden behind compute.
__global__ __launch_bounds__(256)
void gemm_bf16_wmma(const unsigned short* __restrict__ A, int lda,
                    const unsigned short* __restrict__ Bm, int ldb,
                    int K, int mode,
                    unsigned short* __restrict__ t1out, int ld_t1,
                    unsigned short* __restrict__ xbig, int ld_xbig, int mslot, int Dcol,
                    const unsigned short* __restrict__ zprev, int ld_z,
                    const float* __restrict__ bias,
                    float* __restrict__ fout, int ld_f)
{
    __shared__ unsigned short As[128 * LDS_STRIDE];   // [row][k]
    __shared__ unsigned short Bs[128 * LDS_STRIDE];   // [col][k]  (transposed tile)

    const int tid  = threadIdx.x;
    const int lane = tid & 31;
    const int wid  = tid >> 5;
    const int wm   = wid >> 1;        // 0..3 -> 32-row strip
    const int wn   = wid & 1;         // 0..1 -> 64-col strip
    const int p    = lane >> 4;       // K-phase per WMMA 16-bit layout
    const int q    = lane & 15;       // row (A) / col (B) within 16

    const int row0 = blockIdx.y * 128;
    const int col0 = blockIdx.x * 128;

    const v8f vzero = {0.f,0.f,0.f,0.f,0.f,0.f,0.f,0.f};
    v8f acc[2][4];
#pragma unroll
    for (int i = 0; i < 2; ++i)
#pragma unroll
        for (int j = 0; j < 4; ++j) acc[i][j] = vzero;

    // per-thread staging coordinates (2 x 16B chunks for A and for B)
    const int ar0 = tid >> 2;                  // A row for chunk 0 (chunk1: +64)
    const int akc = (tid & 3) << 3;            // A k offset (8 elems)
    const int bk0 = tid >> 4;                  // B k for chunk 0 (chunk1: +16)
    const int bcc = (tid & 15) << 3;           // B col offset (8 elems)

    usv8 aReg[2], bReg[2];
    auto load_tiles = [&](int k0) {
#pragma unroll
        for (int i = 0; i < 2; ++i) {
            aReg[i] = *(const usv8*)(A  + (size_t)(row0 + ar0 + i * 64) * lda + (k0 + akc));
            bReg[i] = *(const usv8*)(Bm + (size_t)(k0 + bk0 + i * 16) * ldb + (col0 + bcc));
        }
    };

    const int ksteps = K >> 5;
    load_tiles(0);
    for (int kt = 0; kt < ksteps; ++kt) {
        // ---- commit staged registers to LDS ----
#pragma unroll
        for (int i = 0; i < 2; ++i) {
            *(usv8*)(&As[(ar0 + i * 64) * LDS_STRIDE + akc]) = aReg[i];
            const int k = bk0 + i * 16;
#pragma unroll
            for (int j = 0; j < 8; ++j)
                Bs[(bcc + j) * LDS_STRIDE + k] = bReg[i][j];
        }
        // ---- kick off next tile's global loads (hidden behind WMMA work) ----
        if (kt + 1 < ksteps) {
            load_tiles((kt + 1) << 5);
            if (kt + 2 < ksteps) {  // gfx1250 global_prefetch_b8 for K tile kt+2
                __builtin_prefetch(A  + (size_t)(row0 + ar0) * lda + ((kt + 2) << 5) + akc, 0, 0);
                __builtin_prefetch(Bm + (size_t)(((kt + 2) << 5) + bk0) * ldb + col0 + bcc, 0, 0);
            }
        }
        __syncthreads();

        // fragment loads: contiguous 2x ds_read_b128 each (16-bit WMMA layout)
        FragBF af[2], bq[4];
#pragma unroll
        for (int ti = 0; ti < 2; ++ti) {
            const unsigned short* ap = &As[(wm * 32 + ti * 16 + q) * LDS_STRIDE + p * 8];
            af[ti].q[0] = *(const uint4*)(ap);
            af[ti].q[1] = *(const uint4*)(ap + 16);
        }
#pragma unroll
        for (int tj = 0; tj < 4; ++tj) {
            const unsigned short* bp = &Bs[(wn * 64 + tj * 16 + q) * LDS_STRIDE + p * 8];
            bq[tj].q[0] = *(const uint4*)(bp);
            bq[tj].q[1] = *(const uint4*)(bp + 16);
        }
#pragma unroll
        for (int ti = 0; ti < 2; ++ti)
#pragma unroll
            for (int tj = 0; tj < 4; ++tj)
                acc[ti][tj] = __builtin_amdgcn_wmma_f32_16x16x32_bf16(
                    false, af[ti].v, false, bq[tj].v, (short)0, acc[ti][tj], false, false);
        __syncthreads();
    }

    // ---- fused epilogues ----
#pragma unroll
    for (int ti = 0; ti < 2; ++ti) {
#pragma unroll
        for (int tj = 0; tj < 4; ++tj) {
#pragma unroll
            for (int v = 0; v < 8; ++v) {
                float val = acc[ti][tj][v];
                const int R  = row0 + wm * 32 + ti * 16 + v + 8 * p;
                const int Cc = col0 + wn * 64 + tj * 16 + q;
                if (mode <= MODE_DIFF2) {
                    if (mode == MODE_DIFF2)
                        val = 2.0f * val - bf2f(zprev[(size_t)R * ld_z + Cc]);
                    const unsigned short bb = f2bf(val);
                    if (mode == MODE_DIFF1)
                        t1out[(size_t)R * ld_t1 + Cc] = bb;
                    // scatter into stacked features: row = b*N + n, col = d*M + mslot
                    const unsigned int b = (unsigned int)Cc / (unsigned int)Dcol;
                    const int d = Cc - (int)b * Dcol;
                    const size_t xr = (size_t)b * 1024 + (size_t)R;
                    xbig[xr * (size_t)ld_xbig + (size_t)(d * 5 + mslot)] = bb;
                } else {
                    val += bias[Cc];
                    val = (mode == MODE_GATE) ? 1.0f / (1.0f + __expf(-val)) : tanhf(val);
                    fout[(size_t)R * ld_f + Cc] = val;
                }
            }
        }
    }
}

// Build Z = concat(x, h or r*h) in GEMM layout [N, B*D] (bf16) and fill Xbig slot m=0.
__global__ void pack_z_kernel(const float* __restrict__ x, int Din,
                              const float* __restrict__ h,
                              const float* __restrict__ rgate,
                              unsigned short* __restrict__ zpack, int BD, int Dcol,
                              unsigned short* __restrict__ xbig, int ld_xbig)
{
    int idx = blockIdx.x * blockDim.x + threadIdx.x;
    const int total = 64 * 1024 * Dcol;
    if (idx >= total) return;
    const int b   = idx / (1024 * Dcol);
    const int rem = idx - b * (1024 * Dcol);
    const int n   = rem / Dcol;
    const int d   = rem - n * Dcol;
    const int row = b * 1024 + n;
    float val;
    if (d < Din) {
        val = x[(size_t)row * Din + d];
    } else {
        const int j = d - Din;
        val = h[(size_t)row * 128 + j];
        if (rgate) val *= rgate[(size_t)row * 256 + j];   // r gate
    }
    const unsigned short bb = f2bf(val);
    zpack[(size_t)n * BD + (size_t)(b * Dcol + d)] = bb;
    xbig[(size_t)row * ld_xbig + (size_t)(d * 5)]  = bb;  // m = 0 slot
}

__global__ void zero_pad_xbig_kernel(unsigned short* __restrict__ xbig, int dm, int dmp)
{
    int i = blockIdx.x * blockDim.x + threadIdx.x;
    const int padc = dmp - dm;
    const int total = 64 * 1024 * padc;
    if (i >= total) return;
    const int r = i / padc;
    const int c = dm + (i - r * padc);
    xbig[(size_t)r * dmp + c] = 0;
}

__global__ void f32_to_bf16_kernel(const float* __restrict__ in, unsigned short* __restrict__ out, int n)
{
    int i = blockIdx.x * blockDim.x + threadIdx.x;
    if (i < n) out[i] = f2bf(in[i]);
}

__global__ void conv_w_kernel(const float* __restrict__ w, unsigned short* __restrict__ wb,
                              int rows, int rowsPad, int cols)
{
    int i = blockIdx.x * blockDim.x + threadIdx.x;
    const int total = rowsPad * cols;
    if (i >= total) return;
    const int r = i / cols;
    const int c = i - r * cols;
    wb[i] = (r < rows) ? f2bf(w[(size_t)r * cols + c]) : (unsigned short)0;
}

// h_new = u*h + (1-u)*c ; write to h and to this timestep's output slot.
__global__ void hnew_kernel(const float* __restrict__ gates, const float* __restrict__ cand,
                            float* __restrict__ h, float* __restrict__ outb)
{
    int i = blockIdx.x * blockDim.x + threadIdx.x;
    if (i >= 64 * 1024 * 128) return;
    const int row = i >> 7;
    const int j   = i & 127;
    const float u  = gates[(size_t)row * 256 + 128 + j];
    const float c  = cand[i];
    const float hn = u * h[i] + (1.0f - u) * c;
    h[i]    = hn;
    outb[i] = hn;
}

extern "C" void kernel_launch(void* const* d_in, const int* in_sizes, int n_in,
                              void* d_out, int out_size, void* d_ws, size_t ws_size,
                              hipStream_t stream)
{
    (void)in_sizes; (void)n_in; (void)out_size; (void)ws_size;
    const float* inputs   = (const float*)d_in[0];
    const float* supports = (const float*)d_in[1];
    const float* init_h   = (const float*)d_in[2];
    const float* Wg0 = (const float*)d_in[3];
    const float* bg0 = (const float*)d_in[4];
    const float* Wc0 = (const float*)d_in[5];
    const float* bc0 = (const float*)d_in[6];
    const float* Wg1 = (const float*)d_in[7];
    const float* bg1 = (const float*)d_in[8];
    const float* Wc1 = (const float*)d_in[9];
    const float* bc1 = (const float*)d_in[10];

    constexpr int NV = 1024, BB = 64, HH = 128, TT = 12;
    constexpr int D0 = 130, BD0 = BB * D0, DM0 = 650, DM0P = 672;  // 672 = 21*32
    constexpr int D1 = 256, BD1 = BB * D1, DM1 = 1280;             // 1280 = 40*32
    constexpr size_t BNH = (size_t)BB * NV * HH;

    // ---- carve workspace ----
    char* ws = (char*)d_ws;
    size_t off = 0;
    auto carve = [&](size_t bytes) -> char* {
        char* p = ws + off;
        off += (bytes + 255) & ~(size_t)255;
        return p;
    };
    unsigned short* supbf = (unsigned short*)carve(2ull * NV * NV * 2);
    unsigned short* wg0b  = (unsigned short*)carve((size_t)DM0P * 256 * 2);
    unsigned short* wc0b  = (unsigned short*)carve((size_t)DM0P * 128 * 2);
    unsigned short* wg1b  = (unsigned short*)carve((size_t)DM1 * 256 * 2);
    unsigned short* wc1b  = (unsigned short*)carve((size_t)DM1 * 128 * 2);
    unsigned short* zpack = (unsigned short*)carve((size_t)NV * BD1 * 2);
    unsigned short* t1    = (unsigned short*)carve((size_t)NV * BD1 * 2);
    unsigned short* xbig  = (unsigned short*)carve((size_t)BB * NV * DM1 * 2);
    float* gates = (float*)carve((size_t)BB * NV * 256 * 4);
    float* cand  = (float*)carve((size_t)BB * NV * 128 * 4);
    float* h0    = (float*)carve(BNH * 4);
    float* h1    = (float*)carve(BNH * 4);

    float* out_hidden = (float*)d_out;            // [2, B, N*H]
    float* out_seq    = out_hidden + 2 * BNH;     // [T, B, N*H]

    // ---- per-launch setup (deterministic) ----
    { int n = 2 * NV * NV; f32_to_bf16_kernel<<<(n + 255) / 256, 256, 0, stream>>>(supports, supbf, n); }
    { int t = DM0P * 256;  conv_w_kernel<<<(t + 255) / 256, 256, 0, stream>>>(Wg0, wg0b, DM0, DM0P, 256); }
    { int t = DM0P * 128;  conv_w_kernel<<<(t + 255) / 256, 256, 0, stream>>>(Wc0, wc0b, DM0, DM0P, 128); }
    { int t = DM1 * 256;   conv_w_kernel<<<(t + 255) / 256, 256, 0, stream>>>(Wg1, wg1b, DM1, DM1, 256); }
    { int t = DM1 * 128;   conv_w_kernel<<<(t + 255) / 256, 256, 0, stream>>>(Wc1, wc1b, DM1, DM1, 128); }
    { int t = BB * NV * (DM0P - DM0);
      zero_pad_xbig_kernel<<<(t + 255) / 256, 256, 0, stream>>>(xbig, DM0, DM0P); }
    hipMemcpyAsync(h0, init_h,        BNH * 4, hipMemcpyDeviceToDevice, stream);
    hipMemcpyAsync(h1, init_h + BNH,  BNH * 4, hipMemcpyDeviceToDevice, stream);

    for (int layer = 0; layer < 2; ++layer) {
        const int Din  = (layer == 0) ? 2   : HH;
        const int D    = (layer == 0) ? D0  : D1;
        const int BD   = (layer == 0) ? BD0 : BD1;
        const int DMP  = (layer == 0) ? DM0P : DM1;
        const unsigned short* wgb = (layer == 0) ? wg0b : wg1b;
        const unsigned short* wcb = (layer == 0) ? wc0b : wc1b;
        const float* bg = (layer == 0) ? bg0 : bg1;
        const float* bc = (layer == 0) ? bc0 : bc1;
        float* h = (layer == 0) ? h0 : h1;

        for (int t = 0; t < TT; ++t) {
            const float* x = (layer == 0) ? (inputs + (size_t)t * BB * NV * 2)
                                          : (out_seq + (size_t)t * BNH);
            float* outb = out_seq + (size_t)t * BNH;
            const dim3 gdiff(BD / 128, NV / 128), blk(256);

            // ---- gate path: Z = [x, h] ----
            { int tot = BB * NV * D;
              pack_z_kernel<<<(tot + 255) / 256, 256, 0, stream>>>(x, Din, h, nullptr, zpack, BD, D, xbig, DMP); }
            for (int s = 0; s < 2; ++s) {
                const unsigned short* S = supbf + (size_t)s * NV * NV;
                gemm_bf16_wmma<<<gdiff, blk, 0, stream>>>(S, NV, zpack, BD, NV, MODE_DIFF1,
                    t1, BD, xbig, DMP, 1 + 2 * s, D, nullptr, 0, nullptr, nullptr, 0);
                gemm_bf16_wmma<<<gdiff, blk, 0, stream>>>(S, NV, t1, BD, NV, MODE_DIFF2,
                    nullptr, 0, xbig, DMP, 2 + 2 * s, D, zpack, BD, nullptr, nullptr, 0);
            }
            { dim3 g(256 / 128, (BB * NV) / 128);
              gemm_bf16_wmma<<<g, blk, 0, stream>>>(xbig, DMP, wgb, 256, DMP, MODE_GATE,
                  nullptr, 0, nullptr, 0, 0, 1, nullptr, 0, bg, gates, 256); }

            // ---- candidate path: Z = [x, r*h] ----
            { int tot = BB * NV * D;
              pack_z_kernel<<<(tot + 255) / 256, 256, 0, stream>>>(x, Din, h, gates, zpack, BD, D, xbig, DMP); }
            for (int s = 0; s < 2; ++s) {
                const unsigned short* S = supbf + (size_t)s * NV * NV;
                gemm_bf16_wmma<<<gdiff, blk, 0, stream>>>(S, NV, zpack, BD, NV, MODE_DIFF1,
                    t1, BD, xbig, DMP, 1 + 2 * s, D, nullptr, 0, nullptr, nullptr, 0);
                gemm_bf16_wmma<<<gdiff, blk, 0, stream>>>(S, NV, t1, BD, NV, MODE_DIFF2,
                    nullptr, 0, xbig, DMP, 2 + 2 * s, D, zpack, BD, nullptr, nullptr, 0);
            }
            { dim3 g(128 / 128, (BB * NV) / 128);
              gemm_bf16_wmma<<<g, blk, 0, stream>>>(xbig, DMP, wcb, 128, DMP, MODE_CAND,
                  nullptr, 0, nullptr, 0, 0, 1, nullptr, 0, bc, cand, 128); }

            // ---- GRU update ----
            { int tot = (int)BNH;
              hnew_kernel<<<(tot + 255) / 256, 256, 0, stream>>>(gates, cand, h, outb); }
        }
        hipMemcpyAsync(out_hidden + (size_t)layer * BNH, h, BNH * 4, hipMemcpyDeviceToDevice, stream);
    }
}
